// ESIMModel_38611755991367
// MI455X (gfx1250) — matrix operand order repaired and checked by reading, not compiled, not run on hardware
//
#include <hip/hip_runtime.h>
#include <hip/hip_bf16.h>

// ---------------------------------------------------------------------------
// ESIM on gfx1250 (MI455X). All GEMMs via v_wmma_f32_16x16x32_f16.
// B=64, T=128, H=300 (gates padded to 320 -> N'=1280), EMB=300->320,
// layer-2 K=2400->2432. All B operands kept N-major so WMMA B fragments are
// single contiguous 32B loads per lane.
// ---------------------------------------------------------------------------

typedef __attribute__((ext_vector_type(16))) _Float16 v16h;
typedef __attribute__((ext_vector_type(8)))  _Float16 v8h;
typedef __attribute__((ext_vector_type(8)))  float    v8f;

#define BATCH 64
#define TT    128
#define NP    1280   // 4 gates * 320
#define GP    320    // padded gate width
#define KE    320    // padded emb / hidden K
#define K2    2432   // padded 8H
#define AD    608    // padded 2H (attention feature dim)

__device__ inline float sigf(float x) { return 1.0f / (1.0f + expf(-x)); }

__device__ inline v8f wmma32(v16h a, v16h b, v8f c) {
  return __builtin_amdgcn_wmma_f32_16x16x32_f16(
      false, a, false, b, (short)0, c, false, false);
}

// A fragment: 16x32 f16 tile, row-major, leading dim lda (halfs).
// lanes 0-15: M=lane, K 0..15; lanes 16-31: M=lane-16, K 16..31
__device__ inline v16h load_a_frag(const _Float16* base, int lda) {
  int lane = threadIdx.x & 31;
  int m  = lane & 15;
  int kb = (lane >> 4) * 16;
  const _Float16* p0 = base + (size_t)m * lda + kb;
  v8h lo = *(const v8h*)(p0);
  v8h hi = *(const v8h*)(p0 + 8);
  v16h r;
#pragma unroll
  for (int i = 0; i < 8; i++) { r[i] = lo[i]; r[8 + i] = hi[i]; }
  return r;
}

// B fragment: 32x16 (K x N) where the matrix is stored N-major:
// element (k,n) = baseT[n*ldt + k]. One contiguous 32B (v16h) load per lane.
// lanes 0-15: K=0..15; lanes 16-31: K=16..31; N = lane & 15.
__device__ inline v16h load_b_frag_nk(const _Float16* baseT, int ldt) {
  int lane = threadIdx.x & 31;
  int n  = lane & 15;
  int kb = (lane >> 4) * 16;
  return *(const v16h*)(baseT + (size_t)n * ldt + kb);
}

// ---------------------------------------------------------------------------
// Weight prep: W (4H x K) f32 -> Wt N-MAJOR (1280 x Kpad) f16,
// gate g occupies rows [320g, 320g+300).
// ---------------------------------------------------------------------------
__global__ void prep_w(const float* __restrict__ W, _Float16* __restrict__ Wt,
                       int K, int Kpad) {
  int idx = blockIdx.x * blockDim.x + threadIdx.x;
  int total = NP * Kpad;
  if (idx >= total) return;
  int n = idx / Kpad, k = idx % Kpad;
  int g = n / GP, j = n % GP;
  float v = 0.0f;
  if (k < K && j < 300) v = W[(size_t)(g * 300 + j) * K + k];
  Wt[idx] = (_Float16)v;
}

__global__ void prep_b(const float* __restrict__ bsrc, float* __restrict__ bdst) {
  int n = blockIdx.x * blockDim.x + threadIdx.x;
  if (n >= NP) return;
  int g = n / GP, j = n % GP;
  bdst[n] = (j < 300) ? bsrc[g * 300 + j] : 0.0f;
}

// ---------------------------------------------------------------------------
// Embedding gather -> f16 padded (B*T, 320)
// ---------------------------------------------------------------------------
__global__ void embed_gather(const int* __restrict__ tok, const float* __restrict__ emb,
                             _Float16* __restrict__ X) {
  int idx = blockIdx.x * blockDim.x + threadIdx.x;
  int total = BATCH * TT * KE;
  if (idx >= total) return;
  int r = idx / KE, k = idx % KE;
  float v = (k < 300) ? emb[(size_t)tok[r] * 300 + k] : 0.0f;
  X[idx] = (_Float16)v;
}

// ---------------------------------------------------------------------------
// Big GEMM: C(M,1280) f16 = A(M,Kp) f16 @ Wt(1280,Kp N-major) f16 + bias
// block: 256 thr (8 waves), tile 128x128, wave tile 32x64 (2x4 WMMA frags)
// ---------------------------------------------------------------------------
__global__ void __launch_bounds__(256)
gemm_zx(const _Float16* __restrict__ A, const _Float16* __restrict__ Wt,
        const float* __restrict__ bias, _Float16* __restrict__ C, int Kp) {
  __shared__ __align__(32) _Float16 As[128 * 32];   // 8 KB, M-major
  __shared__ __align__(32) _Float16 Bs[128 * 32];   // 8 KB, N-major
  int tid  = threadIdx.x;
  int lane = tid & 31, wid = tid >> 5;
  int m0 = blockIdx.y * 128, n0 = blockIdx.x * 128;
  int wm = (wid >> 1) * 32, wn = (wid & 1) * 64;

  v8f acc[2][4];
#pragma unroll
  for (int mi = 0; mi < 2; mi++)
#pragma unroll
    for (int ni = 0; ni < 4; ni++) acc[mi][ni] = (v8f){0,0,0,0,0,0,0,0};

  int r  = tid >> 1;
  int cb = (tid & 1) * 16;
  for (int k0 = 0; k0 < Kp; k0 += 32) {
    *(v16h*)&As[r * 32 + cb] = *(const v16h*)(A  + (size_t)(m0 + r) * Kp + k0 + cb);
    *(v16h*)&Bs[r * 32 + cb] = *(const v16h*)(Wt + (size_t)(n0 + r) * Kp + k0 + cb);
    __syncthreads();
    v16h bfr[4];
#pragma unroll
    for (int ni = 0; ni < 4; ni++)
      bfr[ni] = load_b_frag_nk(Bs + (wn + ni * 16) * 32, 32);
#pragma unroll
    for (int mi = 0; mi < 2; mi++) {
      v16h a = load_a_frag(As + (wm + mi * 16) * 32, 32);
#pragma unroll
      for (int ni = 0; ni < 4; ni++)
        acc[mi][ni] = wmma32(a, bfr[ni], acc[mi][ni]);
    }
    __syncthreads();
  }
  int rowe = (lane >> 4) * 8, coll = lane & 15;
#pragma unroll
  for (int mi = 0; mi < 2; mi++)
#pragma unroll
    for (int ni = 0; ni < 4; ni++)
#pragma unroll
      for (int e = 0; e < 8; e++) {
        int row = m0 + wm + mi * 16 + e + rowe;
        int col = n0 + wn + ni * 16 + coll;
        C[(size_t)row * NP + col] = (_Float16)(acc[mi][ni][e] + bias[col]);
      }
}

// ---------------------------------------------------------------------------
// Persistent recurrent LSTM: one workgroup (512 thr / 16 waves) per chain.
// h lives in LDS (64x320 f16); c + gate accumulators live in registers in
// WMMA C-fragment layout. Per step: z = Z[t] + h @ Wh'  (WMMA), then gates.
// Wh is N-major -> B fragments are contiguous 32B global loads (L2-resident).
// ---------------------------------------------------------------------------
struct ChainArgs {
  const _Float16* Z[4];
  const _Float16* Wh[4];
  float* H[4];
  int rev[4];
};

__global__ void __launch_bounds__(512) lstm_chain(ChainArgs args) {
  int chain = blockIdx.x;
  const _Float16* Z  = args.Z[chain];
  const _Float16* Wh = args.Wh[chain];
  float* H = args.H[chain];
  int rev  = args.rev[chain];

  __shared__ __align__(32) _Float16 hbuf[BATCH * KE];   // 40 KB
  int tid = threadIdx.x, lane = tid & 31, wid = tid >> 5;
  for (int i = tid; i < BATCH * KE; i += 512) hbuf[i] = (_Float16)0;

  v8f cfr[5];
#pragma unroll
  for (int g = 0; g < 5; g++) cfr[g] = (v8f){0,0,0,0,0,0,0,0};
  __syncthreads();

  int rowe = (lane >> 4) * 8, coll = lane & 15;

#pragma unroll 1
  for (int t = 0; t < TT; t++) {
    int teff = rev ? (TT - 1 - t) : t;
    v8f acc[5][4];
    // init accumulators from precomputed input-projection Z[t] (already + bias)
#pragma unroll
    for (int gi = 0; gi < 5; gi++) {
      int grp = wid * 5 + gi;
      int mt = grp / 20, tr = grp % 20;
#pragma unroll
      for (int gate = 0; gate < 4; gate++) {
        v8f z;
#pragma unroll
        for (int e = 0; e < 8; e++) {
          int b   = mt * 16 + e + rowe;
          int col = gate * GP + tr * 16 + coll;
          z[e] = (float)Z[(size_t)(b * TT + teff) * NP + col];
        }
        acc[gi][gate] = z;
      }
    }
    // z += h @ Wh'  (h: LDS A-frags, Wh': contiguous global B-frags)
#pragma unroll
    for (int gi = 0; gi < 5; gi++) {
      int grp = wid * 5 + gi;
      int mt = grp / 20, tr = grp % 20;
      for (int ks = 0; ks < 10; ks++) {
        v16h a = load_a_frag(hbuf + mt * 16 * KE + ks * 32, KE);
#pragma unroll
        for (int gate = 0; gate < 4; gate++) {
          v16h bb = load_b_frag_nk(Wh + (size_t)(gate * GP + tr * 16) * KE + ks * 32, KE);
          acc[gi][gate] = wmma32(a, bb, acc[gi][gate]);
        }
      }
    }
    __syncthreads();   // all hbuf reads done before rewrite
    // gates: i,f,g,o  ->  c,h  (register-resident c; h -> LDS + global)
#pragma unroll
    for (int gi = 0; gi < 5; gi++) {
      int grp = wid * 5 + gi;
      int mt = grp / 20, tr = grp % 20;
#pragma unroll
      for (int e = 0; e < 8; e++) {
        float zi = acc[gi][0][e], zf = acc[gi][1][e];
        float zg = acc[gi][2][e], zo = acc[gi][3][e];
        float c  = sigf(zf) * cfr[gi][e] + sigf(zi) * tanhf(zg);
        float h  = sigf(zo) * tanhf(c);
        cfr[gi][e] = c;
        int b = mt * 16 + e + rowe;
        int j = tr * 16 + coll;
        hbuf[b * KE + j] = (_Float16)h;
        if (j < 300) H[(size_t)(b * TT + teff) * 300 + j] = h;
      }
    }
    __syncthreads();
  }
}

// ---------------------------------------------------------------------------
// Pack a = [h_fwd, h_bwd] -> f16 (B*T, 608) t-major AND (B, 608, 128) d-major
// ---------------------------------------------------------------------------
__global__ void pack_ab(const float* __restrict__ Hf, const float* __restrict__ Hb,
                        _Float16* __restrict__ out, _Float16* __restrict__ outT) {
  int idx = blockIdx.x * blockDim.x + threadIdx.x;
  int total = BATCH * TT * AD;
  if (idx >= total) return;
  int r = idx / AD, j = idx % AD;
  float v = 0.0f;
  if (j < 300)      v = Hf[(size_t)r * 300 + j];
  else if (j < 600) v = Hb[(size_t)r * 300 + j - 300];
  _Float16 h = (_Float16)v;
  out[idx] = h;
  int b = r >> 7, t = r & (TT - 1);
  outT[((size_t)b * AD + j) * TT + t] = h;
}

// ---------------------------------------------------------------------------
// S[b] = a[b] @ b[b]^T : per-wave 16x16 tile, K=608 (19 wmma steps).
// b^T's B operand is N-major (= b row-major) -> contiguous lane loads.
// ---------------------------------------------------------------------------
__global__ void __launch_bounds__(256)
attn_scores(const _Float16* __restrict__ af, const _Float16* __restrict__ bfm,
            float* __restrict__ S) {
  int gw   = (blockIdx.x * blockDim.x + threadIdx.x) >> 5;
  int lane = threadIdx.x & 31;
  int batch = gw >> 6, tile = gw & 63;
  int mt = tile >> 3, nt = tile & 7;
  const _Float16* A  = af  + (size_t)batch * TT * AD + (size_t)(mt * 16) * AD;
  const _Float16* Bm = bfm + (size_t)batch * TT * AD + (size_t)(nt * 16) * AD;
  v8f acc = (v8f){0,0,0,0,0,0,0,0};
  for (int ks = 0; ks < AD / 32; ks++) {
    v16h a  = load_a_frag(A + ks * 32, AD);
    v16h bb = load_b_frag_nk(Bm + ks * 32, AD);
    acc = wmma32(a, bb, acc);
  }
  int rowe = (lane >> 4) * 8, coll = lane & 15;
#pragma unroll
  for (int e = 0; e < 8; e++) {
    int row = mt * 16 + e + rowe, col = nt * 16 + coll;
    S[(size_t)batch * TT * TT + row * TT + col] = acc[e];
  }
}

// ---------------------------------------------------------------------------
// Softmax: mode 0 over q (rows) -> P[(b,p),q]; mode 1 over p (cols) -> PcT[(b,q),p]
// (max-shifted; identical to Eexp/rowsum of the reference)
// ---------------------------------------------------------------------------
__global__ void softmax_dim(const float* __restrict__ S, _Float16* __restrict__ P,
                            int mode) {
  __shared__ float sd[TT];
  int blk = blockIdx.x;          // b*128 + (p or q)
  int b = blk >> 7, idx2 = blk & 127;
  int i = threadIdx.x;           // over the reduced dim
  const float* base = S + (size_t)b * TT * TT;
  float v = (mode == 0) ? base[idx2 * TT + i] : base[i * TT + idx2];
  sd[i] = v; __syncthreads();
  for (int s = 64; s > 0; s >>= 1) { if (i < s) sd[i] = fmaxf(sd[i], sd[i + s]); __syncthreads(); }
  float mx = sd[0]; __syncthreads();
  float e = expf(v - mx);
  sd[i] = e; __syncthreads();
  for (int s = 64; s > 0; s >>= 1) { if (i < s) sd[i] += sd[i + s]; __syncthreads(); }
  P[(size_t)blk * TT + i] = (_Float16)(e / sd[0]);
}

// ---------------------------------------------------------------------------
// Out[b](128,608) = Prob[b](128,128) @ V[b](128,608)   (a_tilde / b_tilde)
// V passed d-major (b, 608, 128) -> contiguous B-fragment loads.
// ---------------------------------------------------------------------------
__global__ void __launch_bounds__(256)
attn_av(const _Float16* __restrict__ P, const _Float16* __restrict__ VT,
        float* __restrict__ Outp) {
  int gw   = (blockIdx.x * blockDim.x + threadIdx.x) >> 5;
  int lane = threadIdx.x & 31;
  int batch = gw / 304, tile = gw % 304;
  int mt = tile / 38, nt = tile % 38;
  const _Float16* A  = P  + (size_t)batch * TT * TT + (size_t)(mt * 16) * TT;
  const _Float16* Bm = VT + (size_t)batch * AD * TT + (size_t)(nt * 16) * TT;
  v8f acc = (v8f){0,0,0,0,0,0,0,0};
#pragma unroll
  for (int ks = 0; ks < 4; ks++) {
    v16h a  = load_a_frag(A + ks * 32, TT);
    v16h bb = load_b_frag_nk(Bm + ks * 32, TT);
    acc = wmma32(a, bb, acc);
  }
  int rowe = (lane >> 4) * 8, coll = lane & 15;
#pragma unroll
  for (int e = 0; e < 8; e++) {
    int row = mt * 16 + e + rowe, col = nt * 16 + coll;
    Outp[(size_t)batch * TT * AD + (size_t)row * AD + col] = acc[e];
  }
}

// ---------------------------------------------------------------------------
// m = [a, at, a-at, a*at] -> f16 (B*T, 2432), zero-padded
// ---------------------------------------------------------------------------
__global__ void build_m(const _Float16* __restrict__ a, const float* __restrict__ at,
                        _Float16* __restrict__ m) {
  int idx = blockIdx.x * blockDim.x + threadIdx.x;
  int total = BATCH * TT * K2;
  if (idx >= total) return;
  int r = idx / K2, j = idx % K2;
  float v = 0.0f;
  if (j < 2400) {
    int s = j / 600, d = j % 600;
    float av = (float)a[(size_t)r * AD + d];
    float tv = at[(size_t)r * AD + d];
    v = (s == 0) ? av : (s == 1) ? tv : (s == 2) ? (av - tv) : (av * tv);
  }
  m[idx] = (_Float16)v;
}

// ---------------------------------------------------------------------------
// Mean-pool over time of [h_fwd, h_bwd] -> av (64, 600)
// ---------------------------------------------------------------------------
__global__ void pool_avg(const float* __restrict__ Hf, const float* __restrict__ Hb,
                         float* __restrict__ av) {
  int idx = blockIdx.x * blockDim.x + threadIdx.x;
  if (idx >= BATCH * 600) return;
  int b = idx / 600, j = idx % 600;
  const float* src = (j < 300) ? Hf + (size_t)(b * TT) * 300 + j
                               : Hb + (size_t)(b * TT) * 300 + (j - 300);
  float s = 0.0f;
  for (int t = 0; t < TT; t++) s += src[(size_t)t * 300];
  av[idx] = s / (float)TT;
}

// ---------------------------------------------------------------------------
// v = [avp, avp, avh, avp]; logits = tanh(v) @ mlp_w^T + b; loss = -mean logp[tgt]
// ---------------------------------------------------------------------------
__global__ void final_loss(const float* __restrict__ avp, const float* __restrict__ avh,
                           const float* __restrict__ mw, const float* __restrict__ mb,
                           const int* __restrict__ target, float* __restrict__ out) {
  __shared__ float acc[BATCH];
  int b = threadIdx.x;
  float lg[3] = {mb[0], mb[1], mb[2]};
  for (int j = 0; j < 600; j++) {
    float tp = tanhf(avp[b * 600 + j]);
    float th = tanhf(avh[b * 600 + j]);
#pragma unroll
    for (int c = 0; c < 3; c++) {
      const float* w = mw + (size_t)c * 2400;
      lg[c] += tp * (w[j] + w[600 + j] + w[1800 + j]) + th * w[1200 + j];
    }
  }
  float mx = fmaxf(lg[0], fmaxf(lg[1], lg[2]));
  float s = expf(lg[0] - mx) + expf(lg[1] - mx) + expf(lg[2] - mx);
  acc[b] = lg[target[b]] - mx - logf(s);
  __syncthreads();
  if (b == 0) {
    float t = 0.0f;
    for (int i = 0; i < BATCH; i++) t += acc[i];
    out[0] = -t / (float)BATCH;
  }
}

// ---------------------------------------------------------------------------
extern "C" void kernel_launch(void* const* d_in, const int* in_sizes, int n_in,
                              void* d_out, int out_size, void* d_ws, size_t ws_size,
                              hipStream_t stream) {
  (void)in_sizes; (void)n_in; (void)out_size; (void)ws_size;
  const int* pre_data = (const int*)d_in[0];
  const int* hyp_data = (const int*)d_in[1];
  const int* target   = (const int*)d_in[2];
  const float* emb    = (const float*)d_in[3];
  // chains: 0=pre_f 1=pre_r 2=hyp_f 3=hyp_r (layer1), same order ipre/ihyp (layer2)
  const float* wi1[4] = {(const float*)d_in[4],  (const float*)d_in[7],  (const float*)d_in[10], (const float*)d_in[13]};
  const float* wh1[4] = {(const float*)d_in[5],  (const float*)d_in[8],  (const float*)d_in[11], (const float*)d_in[14]};
  const float* bb1[4] = {(const float*)d_in[6],  (const float*)d_in[9],  (const float*)d_in[12], (const float*)d_in[15]};
  const float* wi2[4] = {(const float*)d_in[16], (const float*)d_in[19], (const float*)d_in[22], (const float*)d_in[25]};
  const float* wh2[4] = {(const float*)d_in[17], (const float*)d_in[20], (const float*)d_in[23], (const float*)d_in[26]};
  const float* bb2[4] = {(const float*)d_in[18], (const float*)d_in[21], (const float*)d_in[24], (const float*)d_in[27]};
  const float* mlp_w  = (const float*)d_in[28];
  const float* mlp_b  = (const float*)d_in[29];
  float* out = (float*)d_out;

  char* wsb = (char*)d_ws;
  size_t off = 0;
  auto alloc = [&](size_t bytes) -> void* {
    off = (off + 255) & ~(size_t)255;
    void* p = wsb + off;
    off += bytes;
    return p;
  };

  const size_t M = (size_t)BATCH * TT;   // 8192
  _Float16* Xpre = (_Float16*)alloc(M * KE * 2);
  _Float16* Xhyp = (_Float16*)alloc(M * KE * 2);
  _Float16 *Wi1t[4], *Wh1t[4], *Wi2t[4], *Wh2t[4];
  float *b1t[4], *b2t[4];
  _Float16* Z1[4]; float* H1[4];
  _Float16* Z2[4]; float* H2[4];
  for (int i = 0; i < 4; i++) {
    Wi1t[i] = (_Float16*)alloc((size_t)NP * KE * 2);
    Wh1t[i] = (_Float16*)alloc((size_t)NP * KE * 2);
    Wi2t[i] = (_Float16*)alloc((size_t)NP * K2 * 2);
    Wh2t[i] = (_Float16*)alloc((size_t)NP * KE * 2);
    b1t[i]  = (float*)alloc(NP * 4);
    b2t[i]  = (float*)alloc(NP * 4);
    Z1[i]   = (_Float16*)alloc(M * NP * 2);
    Z2[i]   = (_Float16*)alloc(M * NP * 2);
    H1[i]   = (float*)alloc(M * 300 * 4);
    H2[i]   = (float*)alloc(M * 300 * 4);
  }
  _Float16* afh  = (_Float16*)alloc(M * AD * 2);
  _Float16* bfh  = (_Float16*)alloc(M * AD * 2);
  _Float16* afhT = (_Float16*)alloc(M * AD * 2);
  _Float16* bfhT = (_Float16*)alloc(M * AD * 2);
  float*    S    = (float*)alloc((size_t)BATCH * TT * TT * 4);
  _Float16* P    = (_Float16*)alloc((size_t)BATCH * TT * TT * 2);
  _Float16* PcT  = (_Float16*)alloc((size_t)BATCH * TT * TT * 2);
  float*    atl  = (float*)alloc(M * AD * 4);
  float*    btl  = (float*)alloc(M * AD * 4);
  _Float16* ma   = (_Float16*)alloc(M * K2 * 2);
  _Float16* mb_  = (_Float16*)alloc(M * K2 * 2);
  float*    avp  = (float*)alloc((size_t)BATCH * 600 * 4);
  float*    avh  = (float*)alloc((size_t)BATCH * 600 * 4);

  // ---- weight / bias prep (N-major f16, gate-padded) ----
  for (int i = 0; i < 4; i++) {
    prep_w<<<(NP * KE + 255) / 256, 256, 0, stream>>>(wi1[i], Wi1t[i], 300, KE);
    prep_w<<<(NP * KE + 255) / 256, 256, 0, stream>>>(wh1[i], Wh1t[i], 300, KE);
    prep_w<<<(NP * K2 + 255) / 256, 256, 0, stream>>>(wi2[i], Wi2t[i], 2400, K2);
    prep_w<<<(NP * KE + 255) / 256, 256, 0, stream>>>(wh2[i], Wh2t[i], 300, KE);
    prep_b<<<(NP + 255) / 256, 256, 0, stream>>>(bb1[i], b1t[i]);
    prep_b<<<(NP + 255) / 256, 256, 0, stream>>>(bb2[i], b2t[i]);
  }

  // ---- embedding gather ----
  embed_gather<<<(int)((M * KE + 255) / 256), 256, 0, stream>>>(pre_data, emb, Xpre);
  embed_gather<<<(int)((M * KE + 255) / 256), 256, 0, stream>>>(hyp_data, emb, Xhyp);

  // ---- layer-1 input projections (hoisted out of recurrence) ----
  dim3 gz(NP / 128, (unsigned)(M / 128));
  for (int i = 0; i < 4; i++)
    gemm_zx<<<gz, 256, 0, stream>>>((i < 2) ? Xpre : Xhyp, Wi1t[i], b1t[i], Z1[i], KE);

  // ---- layer-1 recurrence: 4 concurrent persistent chains ----
  ChainArgs ca1;
  for (int i = 0; i < 4; i++) { ca1.Z[i] = Z1[i]; ca1.Wh[i] = Wh1t[i]; ca1.H[i] = H1[i]; ca1.rev[i] = i & 1; }
  lstm_chain<<<4, 512, 0, stream>>>(ca1);

  // ---- attention ----
  pack_ab<<<(int)((M * AD + 255) / 256), 256, 0, stream>>>(H1[0], H1[1], afh, afhT);
  pack_ab<<<(int)((M * AD + 255) / 256), 256, 0, stream>>>(H1[2], H1[3], bfh, bfhT);
  attn_scores<<<BATCH * 64 / 8, 256, 0, stream>>>(afh, bfh, S);
  softmax_dim<<<BATCH * TT, TT, 0, stream>>>(S, P, 0);
  softmax_dim<<<BATCH * TT, TT, 0, stream>>>(S, PcT, 1);
  attn_av<<<BATCH * 304 / 8, 256, 0, stream>>>(P, bfhT, atl);
  attn_av<<<BATCH * 304 / 8, 256, 0, stream>>>(PcT, afhT, btl);
  build_m<<<(int)((M * K2 + 255) / 256), 256, 0, stream>>>(afh, atl, ma);
  build_m<<<(int)((M * K2 + 255) / 256), 256, 0, stream>>>(bfh, btl, mb_);

  // ---- layer-2 input projections (the 189-GFLOP GEMMs) ----
  for (int i = 0; i < 4; i++)
    gemm_zx<<<gz, 256, 0, stream>>>((i < 2) ? ma : mb_, Wi2t[i], b2t[i], Z2[i], K2);

  // ---- layer-2 recurrence ----
  ChainArgs ca2;
  for (int i = 0; i < 4; i++) { ca2.Z[i] = Z2[i]; ca2.Wh[i] = Wh2t[i]; ca2.H[i] = H2[i]; ca2.rev[i] = i & 1; }
  lstm_chain<<<4, 512, 0, stream>>>(ca2);

  // ---- pool + MLP + loss ----
  pool_avg<<<(BATCH * 600 + 255) / 256, 256, 0, stream>>>(H2[0], H2[1], avp);
  pool_avg<<<(BATCH * 600 + 255) / 256, 256, 0, stream>>>(H2[2], H2[3], avh);
  final_loss<<<1, BATCH, 0, stream>>>(avp, avh, mlp_w, mlp_b, target, out);
}
